// DeformableAttention3D_19731079757892
// MI455X (gfx1250) — matrix operand order repaired and checked by reading, not compiled, not run on hardware
//
#include <hip/hip_runtime.h>
#include <hip/hip_bf16.h>

#define Bsz   2
#define Cdim  256
#define Nq    32768
#define Hgrid 512
#define Wgrid 64
#define HEADS 8
#define PTS   4
#define DH    32

// Concatenated bf16 weight buffer rows: w_val[0,256) w_off[256,320) w_attn[320,352) w_out[352,608)
#define WROWS_VAL  0
#define WROWS_OFF  256
#define WROWS_ATTN 320
#define WROWS_OUT  352
#define WROWS_TOT  608

typedef __attribute__((ext_vector_type(16))) __bf16 v16bf;
typedef __attribute__((ext_vector_type(8)))  float  v8f;

union BF16x16 { v16bf v; __bf16 e[16]; uint4 q[2]; };

__device__ __forceinline__ __bf16 f2bf(float f) { return (__bf16)f; }  // native v_cvt

// ---------------------------------------------------------------------------
// Kernel 0: one-shot fp32 -> bf16 conversion of all weight matrices.
// 311 KB result stays L2-resident for every GEMM workgroup.
// ---------------------------------------------------------------------------
__global__ __launch_bounds__(256)
void cvtw_kernel(const float* __restrict__ w_val, const float* __restrict__ w_off,
                 const float* __restrict__ w_attn, const float* __restrict__ w_out,
                 __bf16* __restrict__ wbf)
{
    const int i = blockIdx.x * 256 + threadIdx.x;       // one float4 per thread
    const int total4 = WROWS_TOT * Cdim / 4;
    if (i >= total4) return;
    const int e   = i * 4;
    const int row = e >> 8;
    const float* src;
    if (row < WROWS_OFF)       src = w_val  + e;
    else if (row < WROWS_ATTN) src = w_off  + (e - WROWS_OFF  * Cdim);
    else if (row < WROWS_OUT)  src = w_attn + (e - WROWS_ATTN * Cdim);
    else                       src = w_out  + (e - WROWS_OUT  * Cdim);
    float4 v = *reinterpret_cast<const float4*>(src);
    wbf[e + 0] = f2bf(v.x);
    wbf[e + 1] = f2bf(v.y);
    wbf[e + 2] = f2bf(v.z);
    wbf[e + 3] = f2bf(v.w);
}

// ---------------------------------------------------------------------------
// Shared GEMM tile: 16x16 output, K=256 in 8 WMMA steps.
// A from LDS (2x ds 16B), B from pre-converted bf16 weights (2x global 16B).
// ---------------------------------------------------------------------------
__device__ __forceinline__ v8f gemm_tile(const __bf16 (*As)[264], int arow, int khalf,
                                         const __bf16* __restrict__ wrow)
{
    v8f acc = {};
#pragma unroll
    for (int kk = 0; kk < 8; ++kk) {
        BF16x16 a, bm;
        a.q[0]  = *reinterpret_cast<const uint4*>(&As[arow][kk * 32 + khalf]);
        a.q[1]  = *reinterpret_cast<const uint4*>(&As[arow][kk * 32 + 16 + khalf]);
        bm.q[0] = *reinterpret_cast<const uint4*>(wrow + kk * 32);
        bm.q[1] = *reinterpret_cast<const uint4*>(wrow + kk * 32 + 16);
        acc = __builtin_amdgcn_wmma_f32_16x16x32_bf16(
            false, a.v, false, bm.v, (short)0, acc, false, false);
    }
    return acc;
}

// ---------------------------------------------------------------------------
// Kernel 1: fused projections. 128 threads = 4 waves; 64-row n-tile;
// each wave owns a 16-row M-subtile; straight-line phases (no branchy stores).
// ---------------------------------------------------------------------------
__global__ __launch_bounds__(128)
void proj_kernel(const float* __restrict__ x,
                 const __bf16* __restrict__ wbf,
                 const float* __restrict__ b_off, const float* __restrict__ b_attn,
                 const float* __restrict__ b_val,
                 float* __restrict__ value_t,     // (b, heads, N, dh)
                 float* __restrict__ offbuf,      // (b, N, 64)
                 float* __restrict__ attnbuf)     // (b, N, 32)
{
    __shared__ alignas(16) __bf16 As[64][264];

    const int tid  = threadIdx.x;
    const int lane = tid & 31;
    const int wave = tid >> 5;
    const int b    = blockIdx.x / (Nq / 64);
    const int n0   = (blockIdx.x % (Nq / 64)) * 64;

    // Stage A tile: x[b][c][n0..n0+63] -> As[row][c] with native cvt
    for (int c = tid; c < Cdim; c += 128) {
        const float* xp = x + ((size_t)(b * Cdim + c)) * Nq + n0;
#pragma unroll 4
        for (int r = 0; r < 64; r += 4) {
            float4 v = *reinterpret_cast<const float4*>(xp + r);
            As[r + 0][c] = f2bf(v.x);
            As[r + 1][c] = f2bf(v.y);
            As[r + 2][c] = f2bf(v.z);
            As[r + 3][c] = f2bf(v.w);
        }
    }
    __syncthreads();

    const int mrow  = wave * 16;
    const int arow  = lane & 15;
    const int khalf = (lane < 16) ? 0 : 8;
    const int koff  = (lane < 16) ? 0 : 8;
    const int mhi   = (lane < 16) ? 0 : 8;

    // ---- Phase 1: value projection (16 col tiles) ----
    for (int ct = 0; ct < 16; ++ct) {
        const int col = ct * 16 + (lane & 15);
        const __bf16* wrow = wbf + (size_t)(WROWS_VAL + col) * Cdim + koff;
        v8f acc = gemm_tile(As, arow, khalf, wrow);
        const float bv = b_val[col];
        const int head = col >> 5, d = col & 31;
        float* vp = value_t + (((size_t)(b * HEADS + head)) * Nq + n0 + mrow + mhi) * DH + d;
#pragma unroll
        for (int r = 0; r < 8; ++r) vp[(size_t)r * DH] = acc[r] + bv;
    }

    // ---- Phase 2: offset projection (4 col tiles) ----
    for (int ct = 0; ct < 4; ++ct) {
        const int col = ct * 16 + (lane & 15);
        const __bf16* wrow = wbf + (size_t)(WROWS_OFF + col) * Cdim + koff;
        v8f acc = gemm_tile(As, arow, khalf, wrow);
        const float bv = b_off[col];
        float* op = offbuf + ((size_t)(b * Nq + n0 + mrow + mhi)) * 64 + col;
#pragma unroll
        for (int r = 0; r < 8; ++r) op[(size_t)r * 64] = acc[r] + bv;
    }

    // ---- Phase 3: attention logits (2 col tiles) ----
    for (int ct = 0; ct < 2; ++ct) {
        const int col = ct * 16 + (lane & 15);
        const __bf16* wrow = wbf + (size_t)(WROWS_ATTN + col) * Cdim + koff;
        v8f acc = gemm_tile(As, arow, khalf, wrow);
        const float bv = b_attn[col];
        float* ap = attnbuf + ((size_t)(b * Nq + n0 + mrow + mhi)) * 32 + col;
#pragma unroll
        for (int r = 0; r < 8; ++r) ap[(size_t)r * 32] = acc[r] + bv;
    }
}

// ---------------------------------------------------------------------------
// Kernel 2: softmax(P=4) + bilinear gather; one wave per (b, head, n);
// lane = dh channel -> 128B coalesced value_t rows.
// ---------------------------------------------------------------------------
__global__ __launch_bounds__(256)
void sample_kernel(const float* __restrict__ value_t,
                   const float* __restrict__ offbuf,
                   const float* __restrict__ attnbuf,
                   __bf16* __restrict__ sampled)   // (b, N, heads*dh) bf16
{
    const int wid  = blockIdx.x * 8 + (threadIdx.x >> 5);
    const int lane = threadIdx.x & 31;
    if (wid >= Bsz * HEADS * Nq) return;

    const int b    = wid / (HEADS * Nq);
    const int rr   = wid % (HEADS * Nq);
    const int head = rr / Nq;
    const int n    = rr % Nq;

    const float* lg = attnbuf + ((size_t)(b * Nq + n)) * 32 + head * 4;
    const float l0 = lg[0], l1 = lg[1], l2 = lg[2], l3 = lg[3];
    const float mx = fmaxf(fmaxf(l0, l1), fmaxf(l2, l3));
    const float e0 = __expf(l0 - mx), e1 = __expf(l1 - mx);
    const float e2 = __expf(l2 - mx), e3 = __expf(l3 - mx);
    const float inv = 1.0f / (e0 + e1 + e2 + e3);
    const float ew[4] = { e0 * inv, e1 * inv, e2 * inv, e3 * inv };

    const float* op = offbuf + ((size_t)(b * Nq + n)) * 64 + head * 8;
    const int iy = n / Wgrid, ix = n % Wgrid;
    const float* vb = value_t + ((size_t)(b * HEADS + head)) * (size_t)Nq * DH;

    float acc = 0.f;
#pragma unroll
    for (int p = 0; p < PTS; ++p) {
        const float xim = (float)ix + op[p * 2 + 0];   // loc*W - 0.5 collapsed
        const float yim = (float)iy + op[p * 2 + 1];
        const float fx0 = floorf(xim), fy0 = floorf(yim);
        const float fx = xim - fx0, fy = yim - fy0;
        const int x0 = (int)fx0, y0 = (int)fy0;
        const float wp = ew[p];
#pragma unroll
        for (int dy = 0; dy < 2; ++dy) {
#pragma unroll
            for (int dx = 0; dx < 2; ++dx) {
                const int yi = y0 + dy, xi = x0 + dx;
                if (yi >= 0 && yi < Hgrid && xi >= 0 && xi < Wgrid) {
                    const float wgt = (dy ? fy : 1.f - fy) * (dx ? fx : 1.f - fx);
                    acc += wp * wgt * vb[((size_t)(yi * Wgrid + xi)) * DH + lane];
                }
            }
        }
    }
    sampled[((size_t)(b * Nq + n)) * Cdim + head * DH + lane] = f2bf(acc);
}

// ---------------------------------------------------------------------------
// Kernel 3: output projection; A (sampled) already bf16 row-major -> no LDS.
// Stores transposed to (b, c, n) == reference (b,c,z,h,w) layout.
// ---------------------------------------------------------------------------
__global__ __launch_bounds__(128)
void outproj_kernel(const __bf16* __restrict__ sampled,
                    const __bf16* __restrict__ wbf,
                    const float* __restrict__ b_out,
                    float* __restrict__ out)
{
    const int tid  = threadIdx.x;
    const int lane = tid & 31;
    const int wave = tid >> 5;
    const int b    = blockIdx.x / (Nq / 64);
    const int n0   = (blockIdx.x % (Nq / 64)) * 64;

    const int mrow  = wave * 16;
    const int khalf = (lane < 16) ? 0 : 8;
    const int koff  = (lane < 16) ? 0 : 8;
    const int mhi   = (lane < 16) ? 0 : 8;
    const __bf16* arow = sampled + ((size_t)(b * Nq + n0 + mrow + (lane & 15))) * Cdim;

    for (int ct = 0; ct < 16; ++ct) {
        const int col = ct * 16 + (lane & 15);
        const __bf16* wrow = wbf + (size_t)(WROWS_OUT + col) * Cdim + koff;

        v8f acc = {};
#pragma unroll
        for (int kk = 0; kk < 8; ++kk) {
            BF16x16 a, bm;
            a.q[0]  = *reinterpret_cast<const uint4*>(arow + kk * 32 + khalf);
            a.q[1]  = *reinterpret_cast<const uint4*>(arow + kk * 32 + 16 + khalf);
            bm.q[0] = *reinterpret_cast<const uint4*>(wrow + kk * 32);
            bm.q[1] = *reinterpret_cast<const uint4*>(wrow + kk * 32 + 16);
            acc = __builtin_amdgcn_wmma_f32_16x16x32_bf16(
                false, a.v, false, bm.v, (short)0, acc, false, false);
        }

        const float bv = b_out[col];
        float* outp = out + ((size_t)(b * Cdim + col)) * Nq + n0 + mrow + mhi;
#pragma unroll
        for (int r = 0; r < 8; ++r) outp[r] = acc[r] + bv;
    }
}

// ---------------------------------------------------------------------------
extern "C" void kernel_launch(void* const* d_in, const int* in_sizes, int n_in,
                              void* d_out, int out_size, void* d_ws, size_t ws_size,
                              hipStream_t stream) {
    const float* x      = (const float*)d_in[0];
    const float* w_off  = (const float*)d_in[1];
    const float* b_off  = (const float*)d_in[2];
    const float* w_attn = (const float*)d_in[3];
    const float* b_attn = (const float*)d_in[4];
    const float* w_val  = (const float*)d_in[5];
    const float* b_val  = (const float*)d_in[6];
    const float* w_out  = (const float*)d_in[7];
    const float* b_out  = (const float*)d_in[8];
    float* out = (float*)d_out;

    char* ws = (char*)d_ws;
    float*  value_t = (float*)(ws);                          // 64 MiB
    float*  offbuf  = (float*)(ws + (64ull << 20));          // 16 MiB
    float*  attnbuf = (float*)(ws + (80ull << 20));          //  8 MiB
    __bf16* sampled = (__bf16*)(ws + (88ull << 20));         // 32 MiB
    __bf16* wbf     = (__bf16*)(ws + (120ull << 20));        // 311 KiB

    const int total4 = WROWS_TOT * Cdim / 4;
    cvtw_kernel<<<(total4 + 255) / 256, 256, 0, stream>>>(w_val, w_off, w_attn, w_out, wbf);

    const int ntiles = Bsz * (Nq / 64);                      // 1024
    proj_kernel<<<ntiles, 128, 0, stream>>>(x, wbf, b_off, b_attn, b_val,
                                            value_t, offbuf, attnbuf);

    const int nblocks = (Bsz * HEADS * Nq) / 8;              // 65536
    sample_kernel<<<nblocks, 256, 0, stream>>>(value_t, offbuf, attnbuf, sampled);

    outproj_kernel<<<ntiles, 128, 0, stream>>>(sampled, wbf, b_out, out);
}
